// SelfAttention_75823352644204
// MI455X (gfx1250) — matrix operand order repaired and checked by reading, not compiled, not run on hardware
//
#include <hip/hip_runtime.h>

#define EMBED 1024
#define HEADS 16
#define HDIM  64
#define NBATCH 4
#define SEQ   2048
#define SEQW  (SEQ / 32)  // packed-mask words per row

typedef __attribute__((ext_vector_type(16))) __bf16 v16bf;
typedef __attribute__((ext_vector_type(8)))  float  v8f;
typedef int v4i_vec __attribute__((vector_size(16)));

// ---------------- WMMA operand loaders (CDNA5 wave32 layouts, ISA 7.12.2) ----
// A-matrix 16x32 bf16: lane l holds row l%16; elements 0..7 -> K = 8*(l/16)+j,
// elements 8..15 -> K = 16 + 8*(l/16)+j.  Two contiguous 16B chunks per lane.
__device__ __forceinline__ v16bf load_a_vec(const __bf16* base, int stride,
                                            int lane) {
  const __bf16* r = base + (size_t)(lane & 15) * stride + ((lane >> 4) * 8);
  union { uint4 u[2]; v16bf v; } c;
  c.u[0] = *(const uint4*)r;
  c.u[1] = *(const uint4*)(r + 16);
  return c.v;
}

template <typename T>
__device__ __forceinline__ v16bf load_a_cvt(const T* base, int stride, int lane) {
  const int row = lane & 15, kh = (lane >> 4) * 8;
  const T* r = base + (size_t)row * stride + kh;
  v16bf a;
#pragma unroll
  for (int j = 0; j < 8; ++j) {
    a[j]     = (__bf16)(float)r[j];
    a[j + 8] = (__bf16)(float)r[16 + j];
  }
  return a;
}

// B-matrix 32x16: lane l holds column N = l%16; elements 0..7 -> K = 8*(l/16)+j,
// elements 8..15 -> K = 16 + 8*(l/16)+j.  Source row-major [N][K]: two 16B chunks.
__device__ __forceinline__ v16bf load_b_vec(const __bf16* base, int stride,
                                            int lane) {
  const __bf16* r = base + (size_t)(lane & 15) * stride + (lane >> 4) * 8;
  union { uint4 u[2]; v16bf v; } c;
  c.u[0] = *(const uint4*)r;
  c.u[1] = *(const uint4*)(r + 16);
  return c.v;
}

// Same layout but fp32 source with convert (for weight matrices).
__device__ __forceinline__ v16bf load_b_cvt(const float* base, int stride,
                                            int lane) {
  const float* r = base + (size_t)(lane & 15) * stride + (lane >> 4) * 8;
  v16bf b;
#pragma unroll
  for (int j = 0; j < 8; ++j) { b[j] = (__bf16)r[j]; b[j + 8] = (__bf16)r[16 + j]; }
  return b;
}

__device__ __forceinline__ v8f wmma_bf16(v16bf a, v16bf b, v8f c) {
  return __builtin_amdgcn_wmma_f32_16x16x32_bf16(false, a, false, b, (short)0, c,
                                                 false, false);
}

// ---------------- CDNA5 async global->LDS staging ---------------------------
__device__ __forceinline__ void stage16(const __bf16* g, __bf16* l) {
#if __has_builtin(__builtin_amdgcn_global_load_async_to_lds_b128)
  __builtin_amdgcn_global_load_async_to_lds_b128(
      (__attribute__((address_space(1))) v4i_vec*)(g),
      (__attribute__((address_space(3))) v4i_vec*)(l), 0, 0);
#else
  *(uint4*)l = *(const uint4*)g;
#endif
}

template <int N>
__device__ __forceinline__ void wait_async() {
#if __has_builtin(__builtin_amdgcn_s_wait_asynccnt)
  __builtin_amdgcn_s_wait_asynccnt(N);
#else
  if (N == 0) asm volatile("s_wait_asynccnt 0" ::: "memory");
  else        asm volatile("s_wait_asynccnt 4" ::: "memory");
#endif
}

// ---------------- Kernel 0: bit-pack the attention mask ---------------------
__global__ __launch_bounds__(256) void pack_mask_kernel(
    const int* __restrict__ mask, unsigned* __restrict__ pm) {
  const size_t gid = (size_t)blockIdx.x * 256 + threadIdx.x;
  const int pred = mask[gid] != 0;
#if __has_builtin(__builtin_amdgcn_ballot_w32)
  const unsigned word = __builtin_amdgcn_ballot_w32(pred);
#else
  const unsigned word = (unsigned)__ballot(pred);
#endif
  if ((threadIdx.x & 31) == 0) pm[gid >> 5] = word;
}

// ---------------- Kernel 1: per-head input projection  y = x @ W^T ----------
// TRANS=false: y[n][h][s][d]   (Q/K).  TRANS=true: y[n][h][d][s]  (V).
template <bool TRANS>
__global__ __launch_bounds__(128) void proj_kernel(const float* __restrict__ x,
                                                   const float* __restrict__ W,
                                                   __bf16* __restrict__ y) {
  const int lane = threadIdx.x & 31;
  const int wave = threadIdx.x >> 5;
  const int TPS  = SEQ / 16;
  const int tile = blockIdx.x;
  const int s0 = (tile % TPS) * 16;
  const int h  = (tile / TPS) % HEADS;
  const int n  = tile / (TPS * HEADS);
  const int ec = wave * 16;

  const float* xbase = x + ((size_t)n * SEQ + s0) * EMBED + h * HDIM;

  v8f c = {};
#pragma unroll
  for (int kc = 0; kc < HDIM; kc += 32) {
    v16bf a = load_a_cvt(xbase + kc, EMBED, lane);
    v16bf b = load_b_cvt(W + (size_t)ec * HDIM + kc, HDIM, lane);
    c = wmma_bf16(a, b, c);
  }

  const int hw = lane >> 4, ncol = lane & 15;
  __bf16* ybase = y + (size_t)(n * HEADS + h) * SEQ * HDIM;
#pragma unroll
  for (int v = 0; v < 8; ++v) {
    const int srow = s0 + v + 8 * hw;
    if (TRANS)
      ybase[(size_t)(ec + ncol) * SEQ + srow] = (__bf16)c[v];
    else
      ybase[(size_t)srow * HDIM + ec + ncol] = (__bf16)c[v];
  }
}

// ---------------- Kernel 2: flash attention over one (n,h,64-query block) ---
__device__ __forceinline__ void stage_tiles(const __bf16* kbase,
                                            const __bf16* vtbase, int kt,
                                            __bf16* ldsK, __bf16* ldsVt,
                                            int tid) {
#pragma unroll
  for (int i = 0; i < 2; ++i) {
    const int idx = tid + i * 128;  // 0..255 uint4 slots
    stage16(kbase + (size_t)kt * HDIM + idx * 8, ldsK + idx * 8);
    const int row = idx >> 2, c4 = (idx & 3) * 8;
    stage16(vtbase + (size_t)row * SEQ + kt + c4, ldsVt + row * 32 + c4);
  }
}

__device__ __forceinline__ void attn_tile(
    int kt, const __bf16* ldsK, const __bf16* ldsVt, __bf16* pst,
    const unsigned* __restrict__ pmb, int q0, int lane, v16bf qa0, v16bf qa1,
    v8f& o0, v8f& o1, v8f& o2, v8f& o3, float* mrow, float* lrow) {
  const int hw   = lane >> 4;
  const int ncol = lane & 15;
  const float scale = 0.03125f;  // 1/sqrt(EMBED)

  // ---- scores S = Q K^T : two 16x16 f32 tiles
  v8f s0t = {}, s1t = {};
#pragma unroll
  for (int dc = 0; dc < HDIM; dc += 32) {
    v16bf a  = dc ? qa1 : qa0;
    v16bf b0 = load_b_vec(ldsK + 0 * HDIM + dc, HDIM, lane);
    v16bf b1 = load_b_vec(ldsK + 16 * HDIM + dc, HDIM, lane);
    s0t = wmma_bf16(a, b0, s0t);
    s1t = wmma_bf16(a, b1, s1t);
  }

  // ---- masked online softmax (bit-packed mask; one word per row per k-tile)
  float p0[8], p1[8], corr[8];
#pragma unroll
  for (int v = 0; v < 8; ++v) {
    const int qr = q0 + v + 8 * hw;
    const unsigned w = pmb[(size_t)qr * SEQW + (kt >> 5)];
    const float l0 = ((w >> ncol) & 1u)        ? s0t[v] * scale : -1.0e30f;
    const float l1 = ((w >> (16 + ncol)) & 1u) ? s1t[v] * scale : -1.0e30f;
    float t = fmaxf(l0, l1);
#pragma unroll
    for (int m = 1; m <= 8; m <<= 1) t = fmaxf(t, __shfl_xor(t, m, 32));
    const float nm = fmaxf(mrow[v], t);
    corr[v] = __expf(mrow[v] - nm);
    p0[v] = __expf(l0 - nm);
    p1[v] = __expf(l1 - nm);
    float rs = p0[v] + p1[v];
#pragma unroll
    for (int m = 1; m <= 8; m <<= 1) rs += __shfl_xor(rs, m, 32);
    lrow[v] = lrow[v] * corr[v] + rs;
    mrow[v] = nm;
  }

  // ---- rescale O, transpose-stage P (C-layout -> A-layout) via LDS
#pragma unroll
  for (int v = 0; v < 8; ++v) {
    o0[v] *= corr[v]; o1[v] *= corr[v]; o2[v] *= corr[v]; o3[v] *= corr[v];
    const int row = v + 8 * hw;
    pst[row * 32 + ncol]      = (__bf16)p0[v];
    pst[row * 32 + 16 + ncol] = (__bf16)p1[v];
  }
  asm volatile("s_wait_dscnt 0" ::: "memory");  // own-wave LDS RAW

  const v16bf pa  = load_a_vec(pst, 32, lane);
  const v16bf vb0 = load_b_vec(ldsVt + 0 * 32,  32, lane);
  const v16bf vb1 = load_b_vec(ldsVt + 16 * 32, 32, lane);
  const v16bf vb2 = load_b_vec(ldsVt + 32 * 32, 32, lane);
  const v16bf vb3 = load_b_vec(ldsVt + 48 * 32, 32, lane);
  o0 = wmma_bf16(pa, vb0, o0);
  o1 = wmma_bf16(pa, vb1, o1);
  o2 = wmma_bf16(pa, vb2, o2);
  o3 = wmma_bf16(pa, vb3, o3);
}

__global__ __launch_bounds__(128) void attn_kernel(const __bf16* __restrict__ qp,
                                                   const __bf16* __restrict__ kp,
                                                   const __bf16* __restrict__ vpt,
                                                   const unsigned* __restrict__ pm,
                                                   __bf16* __restrict__ attn) {
  __shared__ __align__(16) __bf16 ldsK[2][32 * HDIM];   // [kcol][d]  ping-pong
  __shared__ __align__(16) __bf16 ldsVt[2][HDIM * 32];  // [e][kt]    ping-pong
  __shared__ __align__(16) __bf16 ldsP[4][16 * 32];     // per-wave P staging

  const int lane = threadIdx.x & 31;
  const int wave = threadIdx.x >> 5;
  const int tid  = threadIdx.x;

  const int QTB = SEQ / 64;
  const int b  = blockIdx.x;
  const int q0 = (b % QTB) * 64 + wave * 16;
  const int h  = (b / QTB) % HEADS;
  const int n  = b / (QTB * HEADS);

  const __bf16* qbase  = qp  + ((size_t)(n * HEADS + h) * SEQ + q0) * HDIM;
  const __bf16* kbase  = kp  + (size_t)(n * HEADS + h) * SEQ * HDIM;
  const __bf16* vtbase = vpt + (size_t)(n * HEADS + h) * HDIM * SEQ;
  const unsigned* pmb  = pm  + (size_t)n * SEQ * SEQW;

  const v16bf qa0 = load_a_vec(qbase, HDIM, lane);
  const v16bf qa1 = load_a_vec(qbase + 32, HDIM, lane);

  v8f o0 = {}, o1 = {}, o2 = {}, o3 = {};
  float mrow[8], lrow[8];
#pragma unroll
  for (int v = 0; v < 8; ++v) { mrow[v] = -3.0e38f; lrow[v] = 0.0f; }

  __bf16* pst = ldsP[wave];

  // ---- software-pipelined k-tile loop: async-stage t+1 while computing t ----
  stage_tiles(kbase, vtbase, 0, ldsK[0], ldsVt[0], tid);
  int buf = 0;
  for (int kt = 0; kt < SEQ - 32; kt += 32) {
    stage_tiles(kbase, vtbase, kt + 32, ldsK[buf ^ 1], ldsVt[buf ^ 1], tid);
    wait_async<4>();  // drain tile kt's 4 ops; tile kt+1 keeps streaming
    __syncthreads();
    attn_tile(kt, ldsK[buf], ldsVt[buf], pst, pmb, q0, lane, qa0, qa1,
              o0, o1, o2, o3, mrow, lrow);
    __syncthreads();
    buf ^= 1;
  }
  wait_async<0>();
  __syncthreads();
  attn_tile(SEQ - 32, ldsK[buf], ldsVt[buf], pst, pmb, q0, lane, qa0, qa1,
            o0, o1, o2, o3, mrow, lrow);

  // ---- normalize rows and emit bf16 attention output [n][s][h*64+e]
  const int hw = lane >> 4, ncol = lane & 15;
  __bf16* obase = attn + (size_t)n * SEQ * EMBED + h * HDIM;
#pragma unroll
  for (int v = 0; v < 8; ++v) {
    const int qr = q0 + v + 8 * hw;
    const float inv = 1.0f / lrow[v];
    __bf16* orow = obase + (size_t)qr * EMBED;
    orow[ncol]      = (__bf16)(o0[v] * inv);
    orow[16 + ncol] = (__bf16)(o1[v] * inv);
    orow[32 + ncol] = (__bf16)(o2[v] * inv);
    orow[48 + ncol] = (__bf16)(o3[v] * inv);
  }
}

// ---------------- Kernel 3: output projection  out = attn @ Wo^T + bo -------
__global__ __launch_bounds__(128) void outproj_kernel(
    const __bf16* __restrict__ attn, const float* __restrict__ Wo,
    const float* __restrict__ bo, float* __restrict__ out) {
  const int lane = threadIdx.x & 31;
  const int wave = threadIdx.x >> 5;
  const int CTB  = EMBED / 64;
  const int b  = blockIdx.x;
  const int ec = (b % CTB) * 64 + wave * 16;
  const int r0 = (b / CTB) * 16;

  const __bf16* abase = attn + (size_t)r0 * EMBED;
  v8f c = {};
  for (int kc = 0; kc < EMBED; kc += 32) {
    v16bf a  = load_a_vec(abase + kc, EMBED, lane);
    v16bf bb = load_b_cvt(Wo + (size_t)ec * EMBED + kc, EMBED, lane);
    c = wmma_bf16(a, bb, c);
  }

  const int hw = lane >> 4, ncol = lane & 15;
  const float bias = bo[ec + ncol];
#pragma unroll
  for (int v = 0; v < 8; ++v)
    out[(size_t)(r0 + v + 8 * hw) * EMBED + ec + ncol] = c[v] + bias;
}

// ---------------- Host launcher ---------------------------------------------
extern "C" void kernel_launch(void* const* d_in, const int* in_sizes, int n_in,
                              void* d_out, int out_size, void* d_ws,
                              size_t ws_size, hipStream_t stream) {
  const float* query  = (const float*)d_in[0];
  const float* keys   = (const float*)d_in[1];
  const float* values = (const float*)d_in[2];
  const int*   mask   = (const int*)d_in[3];
  const float* Wq = (const float*)d_in[4];
  const float* Wk = (const float*)d_in[5];
  const float* Wv = (const float*)d_in[6];
  const float* Wo = (const float*)d_in[7];
  const float* bo = (const float*)d_in[8];
  float* out = (float*)d_out;

  const size_t np = (size_t)NBATCH * HEADS * SEQ * HDIM;  // elements per tensor
  __bf16* qp   = (__bf16*)d_ws;
  __bf16* kp   = qp + np;
  __bf16* vpt  = kp + np;                       // transposed V projection
  __bf16* attn = vpt + np;
  unsigned* pmask = (unsigned*)(attn + np);     // bit-packed mask (2 MB)

  const dim3 blk(128);

  const int packGrid = (int)((size_t)NBATCH * SEQ * SEQ / 256);
  pack_mask_kernel<<<packGrid, 256, 0, stream>>>(mask, pmask);

  const int projGrid = NBATCH * HEADS * (SEQ / 16);
  proj_kernel<false><<<projGrid, blk, 0, stream>>>(query,  Wq, qp);
  proj_kernel<false><<<projGrid, blk, 0, stream>>>(keys,   Wk, kp);
  proj_kernel<true ><<<projGrid, blk, 0, stream>>>(values, Wv, vpt);

  const int attnGrid = NBATCH * HEADS * (SEQ / 64);
  attn_kernel<<<attnGrid, blk, 0, stream>>>(qp, kp, vpt, pmask, attn);

  const int outGrid = (NBATCH * SEQ / 16) * (EMBED / 64);
  outproj_kernel<<<outGrid, blk, 0, stream>>>(attn, Wo, bo, out);
}
